// EnsembleRSSM_27736898797597
// MI455X (gfx1250) — matrix operand order repaired and checked by reading, not compiled, not run on hardware
//
#include <hip/hip_runtime.h>

// ---------------------------------------------------------------------------
// EnsembleRSSM observe scan for MI455X (gfx1250, wave32, WMMA bf16).
// GEMMs use v_wmma_f32_16x16x32_bf16 with B tiles staged into LDS via the
// CDNA5 async global->LDS path (GLOBAL_LOAD_ASYNC_TO_LDS_B128, ASYNCcnt),
// double-buffered with the last iteration peeled so the steady-state loop
// is branch-free: stage -> s_wait_asynccnt 2 -> barrier -> 4x WMMA -> barrier.
// ---------------------------------------------------------------------------

typedef __bf16 bf16;
typedef __attribute__((ext_vector_type(16))) __bf16 v16bf;
typedef __attribute__((ext_vector_type(8)))  __bf16 v8bf;
typedef __attribute__((ext_vector_type(8)))  float  v8f;

#define BB 128      // batch
#define TT 64       // time steps
#define EMB 1536
#define ACT 12
#define STOCH 32
#define DETER 1024
#define HIDDEN 1024
#define ENS 5

#if defined(__HIP_DEVICE_COMPILE__) && __has_builtin(__builtin_amdgcn_global_load_async_to_lds_b128)
#define HAVE_ASYNC_LDS 1
#else
#define HAVE_ASYNC_LDS 0
#endif

__device__ __forceinline__ float softplusf(float x) {
    return fmaxf(x, 0.f) + log1pf(expf(-fabsf(x)));
}
__device__ __forceinline__ float sigmoidf(float x) {
    return 1.f / (1.f + expf(-x));
}

template <int N>
__device__ __forceinline__ void wait_asynccnt() {
#if defined(__HIP_DEVICE_COMPILE__)
#if __has_builtin(__builtin_amdgcn_s_wait_asynccnt)
    __builtin_amdgcn_s_wait_asynccnt(N);
#else
    asm volatile("s_wait_asynccnt %0" :: "n"(N) : "memory");
#endif
#endif
}

#if HAVE_ASYNC_LDS
// Stage 32 bytes (16 bf16) global -> LDS with no VGPR round-trip.
// ISA: GLOBAL_LOAD_ASYNC_TO_LDS_B128: vdst = per-lane LDS byte offset,
// vaddr = 64-bit global address; INST_OFFSET is added to BOTH addresses,
// so the second 16B chunk reuses the same operands with offset:16.
__device__ __forceinline__ void stage16_async(const bf16* g, bf16* l) {
    unsigned lds_off = (unsigned)(unsigned long long)
        (__attribute__((address_space(3))) bf16*)l;   // generic -> LDS offset
    asm volatile(
        "global_load_async_to_lds_b128 %0, %1, off\n\t"
        "global_load_async_to_lds_b128 %0, %1, off offset:16"
        :: "v"(lds_off), "v"(g) : "memory");
}
#endif

// ---------------------------------------------------------------------------
// Weight prep: f32 [K,N] row-major  ->  bf16 [N,Kpad] (K-contiguous, zero pad)
// ---------------------------------------------------------------------------
__global__ __launch_bounds__(256) void transpose_to_bf16(
    const float* __restrict__ src, bf16* __restrict__ dst,
    int K, int N, int Kpad)
{
    int n = blockIdx.x;
    for (int kk = threadIdx.x; kk < Kpad; kk += 256)
        dst[(long)n * Kpad + kk] = (kk < K) ? (bf16)src[(long)kk * N + n] : (bf16)0.f;
}

// ---------------------------------------------------------------------------
// bf16 WMMA GEMM:  C[128,N] = A[128,K](bf16) * B[K,N] + bias[N]
// Bt is [N,K] (K-contiguous). 128 threads = 4 waves (wave32).
// Block tile BM=64 x BN=64: wave w -> rows 16w..16w+15, 4 N-tiles per wave.
// grid = (N/64, 128/64, Z).
// A frag (16x32): lane L row=L%16, K halves at k0+(L<16?0:8) and +16.
// B frag (32x16): lane L col=L%16, K halves at k0+(L<16?0:8) and +16.
// ---------------------------------------------------------------------------
__global__ __launch_bounds__(128) void gemm_bf16_wmma(
    const bf16* __restrict__ A, long sA,
    const bf16* __restrict__ Bt, long sB,
    const float* __restrict__ bias, long sBias,
    float* __restrict__ C, long sC,
    int N, int K)
{
    const int z = blockIdx.z;
    A    += (long)z * sA;
    Bt   += (long)z * sB;
    bias += (long)z * sBias;
    C    += (long)z * sC;

    const int n0     = blockIdx.x * 64;
    const int m0     = blockIdx.y * 64;
    const int wave   = threadIdx.x >> 5;
    const int lane   = threadIdx.x & 31;
    const int lane15 = lane & 15;
    const int ahalf  = (lane < 16) ? 0 : 8;   // A K-half select
    const int bbase  = (lane < 16) ? 0 : 8;   // B K-half select

    __shared__ __align__(16) bf16 Blds[2][64 * 32];   // 2 x 4KB double buffer

    v8f acc[4] = {};

    const bf16* arow = A + (long)(m0 + wave * 16 + lane15) * K;

    // cooperative B stager: thread t covers col=t>>1, kk=(t&1)*16 (16 bf16)
    const int ccol = threadIdx.x >> 1;         // 0..63
    const int ckk  = (threadIdx.x & 1) * 16;   // 0 or 16
    const bf16* bsrc = Bt + (long)(n0 + ccol) * K + ckk;
    bf16* ldslot[2] = { &Blds[0][ccol * 32 + ckk], &Blds[1][ccol * 32 + ckk] };

    const int ntiles = K >> 5;   // always >= 2 here (K >= 64)

    // one K-step: load A frag + all 4 B frags, then 4 back-to-back WMMAs
    auto compute_tile = [&](int i, const bf16* cur) {
        union { v16bf v; v8bf h[2]; } afrag;
        const bf16* ak = arow + i * 32 + ahalf;
        afrag.h[0] = *reinterpret_cast<const v8bf*>(ak);
        afrag.h[1] = *reinterpret_cast<const v8bf*>(ak + 16);
        union { v16bf v; v8bf h[2]; } bfrag[4];
#pragma unroll
        for (int j = 0; j < 4; ++j) {
            const bf16* bcol = &cur[(j * 16 + lane15) * 32 + bbase];
            bfrag[j].h[0] = *reinterpret_cast<const v8bf*>(bcol);
            bfrag[j].h[1] = *reinterpret_cast<const v8bf*>(bcol + 16);
        }
#pragma unroll
        for (int j = 0; j < 4; ++j)
            acc[j] = __builtin_amdgcn_wmma_f32_16x16x32_bf16(
                false, afrag.v, false, bfrag[j].v, (short)0, acc[j], false, false);
    };

#if HAVE_ASYNC_LDS
    // ---- async double-buffered pipeline, last iteration peeled ----
    stage16_async(bsrc, ldslot[0]);
    int i = 0;
    for (; i < ntiles - 1; ++i) {
        stage16_async(bsrc + (long)(i + 1) * 32, ldslot[(i + 1) & 1]);
        wait_asynccnt<2>();   // tile i landed; tile i+1 still in flight
        __syncthreads();
        compute_tile(i, Blds[i & 1]);
        __syncthreads();      // all reads done before tile i+2 overwrites
    }
    wait_asynccnt<0>();
    __syncthreads();
    compute_tile(i, Blds[i & 1]);
#else
    // ---- synchronous fallback (reg round-trip, single buffer) ----
    for (int i = 0; i < ntiles; ++i) {
        *reinterpret_cast<v8bf*>(ldslot[0]) =
            *reinterpret_cast<const v8bf*>(bsrc + (long)i * 32);
        *reinterpret_cast<v8bf*>(ldslot[0] + 8) =
            *reinterpret_cast<const v8bf*>(bsrc + (long)i * 32 + 8);
        __syncthreads();
        compute_tile(i, Blds[0]);
        __syncthreads();
    }
#endif

    // epilogue: D layout -> VGPR r: lanes0-15 M=r, lanes16-31 M=r+8
#pragma unroll
    for (int j = 0; j < 4; ++j) {
        const int col = n0 + j * 16 + lane15;
        const float bv = bias[col];
#pragma unroll
        for (int r = 0; r < 8; ++r) {
            const int rowc = m0 + wave * 16 + r + ahalf;
            C[(long)rowc * N + col] = acc[j][r] + bv;
        }
    }
}

// ---------------------------------------------------------------------------
// LayerNorm + ELU -> bf16, generic over (b, k) block index.
// ---------------------------------------------------------------------------
__global__ __launch_bounds__(256) void ln_elu_to_bf16(
    const float* __restrict__ in, long in_b, long in_k,
    const float* __restrict__ g, const float* __restrict__ bn, long gk,
    bf16* __restrict__ outp, long out_b, long out_k,
    int nb, int D)
{
    const int blk = blockIdx.x;
    const int b = blk % nb, k = blk / nb;
    in   += (long)b * in_b + (long)k * in_k;
    g    += (long)k * gk;
    bn   += (long)k * gk;
    outp += (long)b * out_b + (long)k * out_k;

    __shared__ float s1[256], s2[256];
    float sum = 0.f, sq = 0.f;
    for (int i = threadIdx.x; i < D; i += 256) {
        float v = in[i]; sum += v; sq += v * v;
    }
    s1[threadIdx.x] = sum; s2[threadIdx.x] = sq;
    __syncthreads();
    for (int off = 128; off > 0; off >>= 1) {
        if (threadIdx.x < off) {
            s1[threadIdx.x] += s1[threadIdx.x + off];
            s2[threadIdx.x] += s2[threadIdx.x + off];
        }
        __syncthreads();
    }
    const float mean = s1[0] / D;
    const float var  = s2[0] / D - mean * mean;
    const float rs   = rsqrtf(var + 1e-5f);
    for (int i = threadIdx.x; i < D; i += 256) {
        float y = (in[i] - mean) * rs * g[i] + bn[i];
        y = (y > 0.f) ? y : expm1f(y);
        outp[i] = (bf16)y;
    }
}

// ---------------------------------------------------------------------------
// Per-step state prep: apply is_first mask, build bf16 GEMM inputs.
// ---------------------------------------------------------------------------
__global__ __launch_bounds__(256) void prep_step(
    const float* __restrict__ action, const unsigned char* __restrict__ is_first,
    float* __restrict__ stoch, float* __restrict__ deter,
    bf16* __restrict__ xin_bf, bf16* __restrict__ xcat_bf, int t)
{
    const int b = blockIdx.x;
    const float m = is_first[b * TT + t] ? 0.f : 1.f;
    for (int i = threadIdx.x; i < DETER; i += 256) {
        float d = deter[b * DETER + i] * m;
        deter[b * DETER + i] = d;
        xcat_bf[(long)b * 2048 + 1024 + i] = (bf16)d;  // GRU input second half
        if (i < 64) {
            float v;
            if (i < STOCH)            { v = stoch[b * STOCH + i] * m; stoch[b * STOCH + i] = v; }
            else if (i < STOCH + ACT) { v = action[((long)b * TT + t) * ACT + (i - STOCH)] * m; }
            else                      { v = 0.f; }   // zero pad K 44 -> 64
            xin_bf[b * 64 + i] = (bf16)v;
        }
    }
}

// ---------------------------------------------------------------------------
// GRU cell: LN over 3072, gates, new deter.
// ---------------------------------------------------------------------------
__global__ __launch_bounds__(256) void gru_step(
    const float* __restrict__ parts, const float* __restrict__ g,
    const float* __restrict__ bn,
    float* __restrict__ deter, bf16* __restrict__ deter_bf,
    bf16* __restrict__ xobs_bf, float* __restrict__ out, int t)
{
    const int b = blockIdx.x;
    const float* p = parts + (long)b * 3072;

    __shared__ float s1[256], s2[256];
    float sum = 0.f, sq = 0.f;
    for (int i = threadIdx.x; i < 3072; i += 256) {
        float v = p[i]; sum += v; sq += v * v;
    }
    s1[threadIdx.x] = sum; s2[threadIdx.x] = sq;
    __syncthreads();
    for (int off = 128; off > 0; off >>= 1) {
        if (threadIdx.x < off) {
            s1[threadIdx.x] += s1[threadIdx.x + off];
            s2[threadIdx.x] += s2[threadIdx.x + off];
        }
        __syncthreads();
    }
    const float mean = s1[0] / 3072.f;
    const float rs   = rsqrtf(s2[0] / 3072.f - mean * mean + 1e-5f);

    float* drow = deter + (long)b * DETER;
    float* orow = out + ((long)b * TT + t) * 1216 + 192;
    for (int i = threadIdx.x; i < DETER; i += 256) {
        const float nr = (p[i]        - mean) * rs * g[i]        + bn[i];
        const float nc = (p[i + 1024] - mean) * rs * g[i + 1024] + bn[i + 1024];
        const float nu = (p[i + 2048] - mean) * rs * g[i + 2048] + bn[i + 2048];
        const float reset  = sigmoidf(nr);
        const float cand   = tanhf(reset * nc);
        const float update = sigmoidf(nu - 1.f);       // UPDATE_BIAS = -1
        const float d = update * cand + (1.f - update) * drow[i];
        drow[i] = d;
        deter_bf[(long)b * DETER + i] = (bf16)d;       // A for ensemble GEMM
        xobs_bf[(long)b * 2560 + i]   = (bf16)d;       // obs concat first half
        orow[i] = d;                                   // out[..., 192:1216]
    }
}

// embed half of the obs concat input
__global__ __launch_bounds__(256) void prep_embed(
    const float* __restrict__ embed, bf16* __restrict__ xobs_bf, int t)
{
    const int b = blockIdx.x;
    for (int j = threadIdx.x; j < EMB; j += 256)
        xobs_bf[(long)b * 2560 + 1024 + j] = (bf16)embed[((long)b * TT + t) * EMB + j];
}

// distribution heads: softplus std, select ensemble member, write out slice
__global__ __launch_bounds__(32) void finalize_step(
    const float* __restrict__ odist, const float* __restrict__ dsout,
    const int* __restrict__ ens_index,
    float* __restrict__ stoch, float* __restrict__ out, int t)
{
    const int b = blockIdx.x;
    const int j = threadIdx.x;              // 0..31
    const int idx = ens_index[t];
    const float om = odist[b * 64 + j];
    const float os = softplusf(odist[b * 64 + 32 + j]) + 0.1f;
    const float pm = dsout[((long)idx * BB + b) * 64 + j];
    const float ps = softplusf(dsout[((long)idx * BB + b) * 64 + 32 + j]) + 0.1f;
    float* o = out + ((long)b * TT + t) * 1216;
    o[j]        = om;   // omean
    o[32 + j]   = os;   // ostd
    o[64 + j]   = om;   // post_stoch
    o[96 + j]   = pm;   // pmean
    o[128 + j]  = ps;   // pstd
    o[160 + j]  = pm;   // prior_stoch
    stoch[b * STOCH + j] = om;
}

__global__ __launch_bounds__(256) void zero_state(
    float* __restrict__ stoch, float* __restrict__ deter)
{
    const int i = blockIdx.x * 256 + threadIdx.x;
    if (i < BB * STOCH)  stoch[i] = 0.f;
    if (i < BB * DETER)  deter[i] = 0.f;
}

// ---------------------------------------------------------------------------
extern "C" void kernel_launch(void* const* d_in, const int* in_sizes, int n_in,
                              void* d_out, int out_size, void* d_ws, size_t ws_size,
                              hipStream_t stream) {
    (void)in_sizes; (void)n_in; (void)out_size; (void)ws_size;

    // inputs in setup_inputs() order
    const float*         embed      = (const float*)d_in[0];
    const float*         action     = (const float*)d_in[1];
    const unsigned char* is_first   = (const unsigned char*)d_in[2];
    const int*           ens_index  = (const int*)d_in[3];
    const float* W_img_in   = (const float*)d_in[4];
    const float* b_img_in   = (const float*)d_in[5];
    const float* g_img_in   = (const float*)d_in[6];
    const float* bn_img_in  = (const float*)d_in[7];
    const float* W_gru      = (const float*)d_in[8];
    const float* b_gru      = (const float*)d_in[9];
    const float* g_gru      = (const float*)d_in[10];
    const float* bn_gru     = (const float*)d_in[11];
    const float* W_img_out  = (const float*)d_in[12];
    const float* b_img_out  = (const float*)d_in[13];
    const float* g_img_out  = (const float*)d_in[14];
    const float* bn_img_out = (const float*)d_in[15];
    const float* W_img_dist = (const float*)d_in[16];
    const float* b_img_dist = (const float*)d_in[17];
    const float* W_obs_out  = (const float*)d_in[18];
    const float* b_obs_out  = (const float*)d_in[19];
    const float* g_obs_out  = (const float*)d_in[20];
    const float* bn_obs_out = (const float*)d_in[21];
    const float* W_obs_dist = (const float*)d_in[22];
    const float* b_obs_dist = (const float*)d_in[23];

    float* out = (float*)d_out;

    // ---- workspace carve-out (≈38 MB) ----
    char* p = (char*)d_ws;
    auto carve = [&](size_t bytes) -> char* {
        char* r = p; p += (bytes + 255) & ~(size_t)255; return r;
    };
    bf16* Wt_img_in   = (bf16*)carve((size_t)1024 * 64   * 2);
    bf16* Wt_gru      = (bf16*)carve((size_t)3072 * 2048 * 2);
    bf16* Wt_img_out  = (bf16*)carve((size_t)5120 * 1024 * 2);
    bf16* Wt_img_dist = (bf16*)carve((size_t)ENS * 64 * 1024 * 2);
    bf16* Wt_obs_out  = (bf16*)carve((size_t)1024 * 2560 * 2);
    bf16* Wt_obs_dist = (bf16*)carve((size_t)64 * 1024 * 2);

    float* stoch    = (float*)carve((size_t)BB * STOCH * 4);
    float* deter    = (float*)carve((size_t)BB * DETER * 4);
    bf16*  xin_bf   = (bf16*) carve((size_t)BB * 64 * 2);
    bf16*  xcat_bf  = (bf16*) carve((size_t)BB * 2048 * 2);
    float* pre1     = (float*)carve((size_t)BB * 1024 * 4);
    float* parts    = (float*)carve((size_t)BB * 3072 * 4);
    bf16*  deter_bf = (bf16*) carve((size_t)BB * 1024 * 2);
    float* hpre     = (float*)carve((size_t)BB * 5120 * 4);
    bf16*  hs_bf    = (bf16*) carve((size_t)ENS * BB * 1024 * 2);
    float* dsout    = (float*)carve((size_t)ENS * BB * 64 * 4);
    bf16*  xobs_bf  = (bf16*) carve((size_t)BB * 2560 * 2);
    float* obs_pre  = (float*)carve((size_t)BB * 1024 * 4);
    bf16*  xo_bf    = (bf16*) carve((size_t)BB * 1024 * 2);
    float* odist    = (float*)carve((size_t)BB * 64 * 4);

    // ---- one-time weight convert/transpose to bf16 [N,K] ----
    transpose_to_bf16<<<1024, 256, 0, stream>>>(W_img_in, Wt_img_in, 44, 1024, 64);
    transpose_to_bf16<<<3072, 256, 0, stream>>>(W_gru, Wt_gru, 2048, 3072, 2048);
    for (int k = 0; k < ENS; ++k) {
        transpose_to_bf16<<<1024, 256, 0, stream>>>(
            W_img_out + (size_t)k * 1024 * 1024, Wt_img_out + (size_t)k * 1024 * 1024,
            1024, 1024, 1024);
        transpose_to_bf16<<<64, 256, 0, stream>>>(
            W_img_dist + (size_t)k * 1024 * 64, Wt_img_dist + (size_t)k * 64 * 1024,
            1024, 64, 1024);
    }
    transpose_to_bf16<<<1024, 256, 0, stream>>>(W_obs_out, Wt_obs_out, 2560, 1024, 2560);
    transpose_to_bf16<<<64, 256, 0, stream>>>(W_obs_dist, Wt_obs_dist, 1024, 64, 1024);

    zero_state<<<(BB * DETER + 255) / 256, 256, 0, stream>>>(stoch, deter);

    auto gemm = [&](const bf16* A, long sA, const bf16* Bt, long sB,
                    const float* bias, long sBias, float* C, long sC,
                    int N, int K, int Z) {
        dim3 g(N / 64 > 0 ? N / 64 : 1, BB / 64, Z);
        gemm_bf16_wmma<<<g, 128, 0, stream>>>(A, sA, Bt, sB, bias, sBias, C, sC, N, K);
    };

    // ---- sequential scan over T ----
    for (int t = 0; t < TT; ++t) {
        prep_step<<<BB, 256, 0, stream>>>(action, is_first, stoch, deter,
                                          xin_bf, xcat_bf, t);
        // x = elu(LN(cat(stoch,a) @ W_img_in + b))
        gemm(xin_bf, 0, Wt_img_in, 0, b_img_in, 0, pre1, 0, 1024, 64, 1);
        ln_elu_to_bf16<<<BB, 256, 0, stream>>>(pre1, 1024, 0, g_img_in, bn_img_in, 0,
                                               xcat_bf, 2048, 0, BB, 1024);
        // GRU: parts = LN(cat(x,deter) @ W_gru + b); gates -> deter
        gemm(xcat_bf, 0, Wt_gru, 0, b_gru, 0, parts, 0, 3072, 2048, 1);
        gru_step<<<BB, 256, 0, stream>>>(parts, g_gru, bn_gru, deter, deter_bf,
                                         xobs_bf, out, t);
        // ensemble heads: hs = elu(LN(deter @ W_img_out[k] + b[k]))
        gemm(deter_bf, 0, Wt_img_out, 0, b_img_out, 0, hpre, 0, 5120, 1024, 1);
        ln_elu_to_bf16<<<ENS * BB, 256, 0, stream>>>(hpre, 5120, 1024,
                                                     g_img_out, bn_img_out, 1024,
                                                     hs_bf, 1024, (long)BB * 1024,
                                                     BB, 1024);
        // ds[k] = hs[k] @ W_img_dist[k] + b[k]   (batched over grid.z)
        gemm(hs_bf, (long)BB * 1024, Wt_img_dist, (long)64 * 1024,
             b_img_dist, 64, dsout, (long)BB * 64, 64, 1024, ENS);
        // obs posterior: xo = elu(LN(cat(deter, embed) @ W_obs_out + b))
        prep_embed<<<BB, 256, 0, stream>>>(embed, xobs_bf, t);
        gemm(xobs_bf, 0, Wt_obs_out, 0, b_obs_out, 0, obs_pre, 0, 1024, 2560, 1);
        ln_elu_to_bf16<<<BB, 256, 0, stream>>>(obs_pre, 1024, 0, g_obs_out, bn_obs_out, 0,
                                               xo_bf, 1024, 0, BB, 1024);
        gemm(xo_bf, 0, Wt_obs_dist, 0, b_obs_dist, 0, odist, 0, 64, 1024, 1);
        finalize_step<<<BB, 32, 0, stream>>>(odist, dsout, ens_index, stoch, out, t);
    }
}
